// ODEModel_39135742001490
// MI455X (gfx1250) — matrix operand order repaired and checked by reading, not compiled, not run on hardware
//
#include <hip/hip_runtime.h>
#include <hip/hip_bf16.h>

typedef __attribute__((ext_vector_type(16))) _Float16 v16h;
typedef __attribute__((ext_vector_type(8)))  _Float16 v8h;
typedef __attribute__((ext_vector_type(8)))  float    v8f;
typedef __attribute__((ext_vector_type(4)))  float    v4f;

// Problem dims (fixed by reference)
constexpr int IN   = 128;   // input/output feature dim
constexpr int HID  = 256;   // hidden dim
constexpr int WAVES = 8;    // waves per workgroup (wave32)
constexpr int TPB  = WAVES * 32;   // 256 threads
constexpr int ROWS = WAVES * 16;   // 128 batch rows per workgroup

// LDS row strides (halves), padded by 8 halves (16B) to spread banks.
constexpr int W1S = IN  + 8;   // 136
constexpr int W2S = HID + 8;   // 264
constexpr int YS  = IN  + 8;   // 136
constexpr int HS  = HID + 8;   // 264

__device__ __forceinline__ v8f wmma16(v16h a, v16h b, v8f c) {
  // v_wmma_f32_16x16x32_f16: D = A(16x32) * B(32x16) + C(16x16 f32)
  return __builtin_amdgcn_wmma_f32_16x16x32_f16(
      /*neg_a=*/false, a, /*neg_b=*/false, b,
      /*c_mod=*/(short)0, c, /*reuse_a=*/false, /*reuse_b=*/false);
}

// A-matrix fragment (16x32 f16): lane L holds row M=L%16; the lane's 16
// halves are two contiguous 8-half chunks at K = 8*hi and K = 16+8*hi.
__device__ __forceinline__ v16h load_fragA(const _Float16* m, int stride,
                                           int row, int kbase, int hi) {
  const _Float16* p = m + row * stride + kbase + 8 * hi;
  v8h a = *(const v8h*)(p);
  v8h b = *(const v8h*)(p + 16);
  return __builtin_shufflevector(a, b, 0, 1, 2, 3, 4, 5, 6, 7,
                                 8, 9, 10, 11, 12, 13, 14, 15);
}

// B-matrix fragment (32x16 f16): lane L holds column N=L%16; the lane's 16
// halves are two contiguous 8-half chunks at K = 8*hi and K = 16+8*hi.
__device__ __forceinline__ v16h load_fragB(const _Float16* m, int stride,
                                           int row, int kbase, int hi) {
  const _Float16* p = m + row * stride + kbase + 8 * hi;
  v8h a = *(const v8h*)(p);
  v8h b = *(const v8h*)(p + 16);
  return __builtin_shufflevector(a, b, 0, 1, 2, 3, 4, 5, 6, 7,
                                 8, 9, 10, 11, 12, 13, 14, 15);
}

__global__ __launch_bounds__(TPB) void node_rk4_wmma(
    const float* __restrict__ x,  const float* __restrict__ W1,
    const float* __restrict__ b1, const float* __restrict__ W2,
    const float* __restrict__ b2, float* __restrict__ out) {
  __shared__ __align__(16) _Float16 sW1[HID * W1S];  // 69632 B
  __shared__ __align__(16) _Float16 sW2[IN * W2S];   // 67584 B
  __shared__ __align__(16) _Float16 sY[ROWS * YS];   // 34816 B  (current state, f16)
  __shared__ __align__(16) _Float16 sH[ROWS * HS];   // 67584 B  (hidden acts, f16)
  __shared__ float sB1[HID];
  __shared__ float sB2[IN];

  const int tid = threadIdx.x;
  // One-time weight/bias staging: f32 global -> f16 LDS (padded strides).
  for (int i = tid; i < HID * IN; i += TPB)
    sW1[(i >> 7) * W1S + (i & 127)] = (_Float16)W1[i];
  for (int i = tid; i < IN * HID; i += TPB)
    sW2[(i >> 8) * W2S + (i & 255)] = (_Float16)W2[i];
  for (int i = tid; i < HID; i += TPB) sB1[i] = b1[i];
  for (int i = tid; i < IN;  i += TPB) sB2[i] = b2[i];
  __syncthreads();

  const int lane = tid & 31;
  const int wave = tid >> 5;
  const int lm   = lane & 15;  // row/col within 16x16 tile
  const int hi   = lane >> 4;  // which half-wave
  const int rowLocal = wave * 16 + lm;                      // batch row in WG
  const size_t rowGlobal = (size_t)blockIdx.x * ROWS + rowLocal;

  // Register-resident per-lane state. Flat index i = jt*8 + r corresponds to
  // feature j = 16*jt + 8*hi + r  (matches the C/D VGPR layout of GEMM2).
  float xr[64], k1[64], k2[64];

  const float* xg = x + rowGlobal * IN + 8 * hi;
  _Float16* yrow = sY + rowLocal * YS + 8 * hi;
#pragma unroll
  for (int jt = 0; jt < 8; ++jt) {
    v4f a = *(const v4f*)(xg + jt * 16);
    v4f b = *(const v4f*)(xg + jt * 16 + 4);
    v8h o;
#pragma unroll
    for (int r = 0; r < 4; ++r) { xr[jt * 8 + r] = a[r]; o[r] = (_Float16)a[r]; }
#pragma unroll
    for (int r = 0; r < 4; ++r) { xr[jt * 8 + 4 + r] = b[r]; o[4 + r] = (_Float16)b[r]; }
    *(v8h*)(yrow + jt * 16) = o;  // y0 = x
  }

#pragma unroll
  for (int st = 0; st < 4; ++st) {
    // ---- GEMM1: h^T = relu(W1 * y^T + b1). D-tile rows = hidden n, cols = batch m.
    // The y B-fragments are identical for every nt tile -> load once per stage.
    v16h yf0 = load_fragB(sY, YS, rowLocal, 0 * 32, hi);
    v16h yf1 = load_fragB(sY, YS, rowLocal, 1 * 32, hi);
    v16h yf2 = load_fragB(sY, YS, rowLocal, 2 * 32, hi);
    v16h yf3 = load_fragB(sY, YS, rowLocal, 3 * 32, hi);

#pragma unroll 4
    for (int nt = 0; nt < 16; ++nt) {
      const _Float16* arow = sW1 + (nt * 16 + lm) * W1S + 8 * hi;
      v16h A0 = load_fragA(sW1, W1S, nt * 16 + lm, 0 * 32, hi);
      v16h A1 = load_fragA(sW1, W1S, nt * 16 + lm, 1 * 32, hi);
      v16h A2 = load_fragA(sW1, W1S, nt * 16 + lm, 2 * 32, hi);
      v16h A3 = load_fragA(sW1, W1S, nt * 16 + lm, 3 * 32, hi);
      (void)arow;
      v8f z = {0.f, 0.f, 0.f, 0.f, 0.f, 0.f, 0.f, 0.f};
      v8f acc0 = wmma16(A0, yf0, z);
      v8f acc1 = wmma16(A1, yf1, z);
      acc0 = wmma16(A2, yf2, acc0);
      acc1 = wmma16(A3, yf3, acc1);
      v8f acc = acc0 + acc1;
      const float* bp = sB1 + nt * 16 + 8 * hi;
      v8h o;
#pragma unroll
      for (int r = 0; r < 8; ++r) {
        float v = acc[r] + bp[r];
        v = v > 0.f ? v : 0.f;  // ReLU
        o[r] = (_Float16)v;
      }
      *(v8h*)(sH + rowLocal * HS + nt * 16 + 8 * hi) = o;
    }

    // ---- GEMM2: k^T = W2 * h^T + b2, fused with the RK4 stage update.
    // The h B-fragments are identical for every jt tile -> load once per stage.
    v16h hf[8];
#pragma unroll
    for (int nt2 = 0; nt2 < 8; ++nt2)
      hf[nt2] = load_fragB(sH, HS, rowLocal, nt2 * 32, hi);

#pragma unroll
    for (int jt = 0; jt < 8; ++jt) {
      const int arowBase = (jt * 16 + lm);
      v8f z = {0.f, 0.f, 0.f, 0.f, 0.f, 0.f, 0.f, 0.f};
      v8f acc0 = z, acc1 = z;
#pragma unroll
      for (int nt2 = 0; nt2 < 8; nt2 += 2) {
        v16h A0 = load_fragA(sW2, W2S, arowBase, nt2 * 32, hi);
        v16h A1 = load_fragA(sW2, W2S, arowBase, (nt2 + 1) * 32, hi);
        acc0 = wmma16(A0, hf[nt2], acc0);
        acc1 = wmma16(A1, hf[nt2 + 1], acc1);
      }
      v8f acc = acc0 + acc1;
      const float* bp = sB2 + jt * 16 + 8 * hi;

      if (st < 3) {
        v8h o;
#pragma unroll
        for (int r = 0; r < 8; ++r) {
          const int i = jt * 8 + r;
          const float k = acc[r] + bp[r];
          float yn;
          if (st == 0) {        // k1; y2 = x + k1/3
            k1[i] = k;
            yn = xr[i] + k * (1.0f / 3.0f);
          } else if (st == 1) { // k2; y3 = x + k2 - k1/3
            k2[i] = k;
            yn = xr[i] + k - k1[i] * (1.0f / 3.0f);
          } else {              // k3; y4 = x + k1 - k2 + k3; fold s = k1+3(k2+k3)
            yn = xr[i] + k1[i] - k2[i] + k;
            k1[i] = k1[i] + 3.0f * (k2[i] + k);
          }
          o[r] = (_Float16)yn;
        }
        *(v8h*)(yrow + jt * 16) = o;  // next stage's y
      } else {
        // k4; out = x + 0.125 * (k1 + 3(k2+k3) + k4)  (s already folded into k1)
        float* og = out + rowGlobal * IN + jt * 16 + 8 * hi;
        v4f o0, o1;
#pragma unroll
        for (int r = 0; r < 4; ++r)
          o0[r] = xr[jt * 8 + r] + 0.125f * (k1[jt * 8 + r] + acc[r] + bp[r]);
#pragma unroll
        for (int r = 0; r < 4; ++r)
          o1[r] = xr[jt * 8 + 4 + r] + 0.125f * (k1[jt * 8 + 4 + r] + acc[4 + r] + bp[4 + r]);
        *(v4f*)(og) = o0;
        *(v4f*)(og + 4) = o1;
      }
    }
  }
}

extern "C" void kernel_launch(void* const* d_in, const int* in_sizes, int n_in,
                              void* d_out, int out_size, void* d_ws, size_t ws_size,
                              hipStream_t stream) {
  (void)n_in; (void)d_ws; (void)ws_size; (void)out_size;
  const float* x  = (const float*)d_in[0];
  const float* W1 = (const float*)d_in[1];
  const float* b1 = (const float*)d_in[2];
  const float* W2 = (const float*)d_in[3];
  const float* b2 = (const float*)d_in[4];
  float* out = (float*)d_out;

  const int batch  = in_sizes[0] / IN;   // 262144
  const int blocks = batch / ROWS;       // 2048 workgroups of 8 waves
  node_rk4_wmma<<<dim3(blocks), dim3(TPB), 0, stream>>>(x, W1, b1, W2, b2, out);
}
